// EncoderWithClassifier_60086592471382
// MI455X (gfx1250) — compile-verified, matrix operand order checked by reading, not run on hardware
//
#include <hip/hip_runtime.h>
#include <cstddef>
#include <cstdint>

// ---------------- model constants ----------------
#define B_   4
#define T_   2048
#define C_   256
#define H_   8
#define HS_  32
#define L_   4
#define FFN_ 256
#define CLSH_ 512
#define NOUT_ 10
#define NTOK_ (B_*T_)          // 8192 rows
#define EPS_ 1e-5f

typedef __bf16 v16bf __attribute__((ext_vector_type(16)));
typedef float  v8f   __attribute__((ext_vector_type(8)));
typedef int    v4i   __attribute__((vector_size(16)));   // pointee type of async b128 builtin
typedef int    v2i   __attribute__((vector_size(8)));    // pointee type of async b64 builtin

#define AS1 __attribute__((address_space(1)))
#define AS3 __attribute__((address_space(3)))

// ---- CDNA5 async global->LDS copy support (probe via __has_builtin) ----
#if __has_builtin(__builtin_amdgcn_global_load_async_to_lds_b128) && __has_builtin(__builtin_amdgcn_s_wait_asynccnt)
#define GEMM_ASYNC 1
#else
#define GEMM_ASYNC 0
#endif
#if __has_builtin(__builtin_amdgcn_global_load_async_to_lds_b64) && __has_builtin(__builtin_amdgcn_s_wait_asynccnt)
#define ATTN_ASYNC 1
#else
#define ATTN_ASYNC 0
#endif

#if GEMM_ASYNC
static __device__ __forceinline__ void async_cp16(const __bf16* g, __bf16* l) {
    __builtin_amdgcn_global_load_async_to_lds_b128(
        (AS1 v4i*)(AS1 void*)g, (AS3 v4i*)(AS3 void*)l, 0, 0);
}
#endif
#if ATTN_ASYNC
static __device__ __forceinline__ void async_cp8(const __bf16* g, __bf16* l) {
    __builtin_amdgcn_global_load_async_to_lds_b64(
        (AS1 v2i*)(AS1 void*)g, (AS3 v2i*)(AS3 void*)l, 0, 0);
}
#endif
#if GEMM_ASYNC || ATTN_ASYNC
static __device__ __forceinline__ void wait_async0() {
    __builtin_amdgcn_s_wait_asynccnt(0);
}
#endif

static __device__ __forceinline__ __bf16 f2bf(float f) {
    unsigned u = __builtin_bit_cast(unsigned, f);
    unsigned r = (u + 0x7fffu + ((u >> 16) & 1u)) >> 16;
    unsigned short s = (unsigned short)r;
    return __builtin_bit_cast(__bf16, s);
}

// Load a 16-element bf16 fragment whose elements live as two contiguous
// 8-element (16B) runs at p[0..7] and p[16..23].  p must be 16B aligned.
static __device__ __forceinline__ v16bf load_frag(const __bf16* p) {
    union { uint4 u[2]; v16bf v; } t;
    t.u[0] = *(const uint4*)p;
    t.u[1] = *(const uint4*)(p + 16);
    return t.v;
}

// ---------------- weight packing: fp32 -> bf16, TRANSPOSED (B^T[N][K]) ----------------
// qkvwT [L][768][C]  row j: 0-255 Q, 256-511 K, 512-767 V ; head=j/32, d=j%32
// projwT[L][C][C] (= Wproj^T), w1T [L][FFN][C], w2T [L][C][FFN]
__global__ __launch_bounds__(256) void pack_weights_k(
    const float* __restrict__ Wq, const float* __restrict__ Wk, const float* __restrict__ Wv,
    const float* __restrict__ Wp, const float* __restrict__ W1, const float* __restrict__ W2,
    __bf16* __restrict__ qkvwT, __bf16* __restrict__ projwT,
    __bf16* __restrict__ w1T,  __bf16* __restrict__ w2T)
{
    const int PER_L = C_*768 + C_*C_ + C_*FFN_ + FFN_*C_;   // 393216
    long id = (long)blockIdx.x * 256 + threadIdx.x;
    int l = (int)(id / PER_L);
    int rem = (int)(id % PER_L);
    if (l >= L_) return;
    if (rem < C_*768) {                       // qkvwT[j][c]
        int j = rem / C_, c = rem % C_;
        int sel = j >> 8, jj = j & 255, hh = jj >> 5, dd = jj & 31;
        const float* W = (sel == 0) ? Wq : (sel == 1) ? Wk : Wv;
        qkvwT[(long)l*C_*768 + rem] = f2bf(W[(((long)l*H_ + hh)*C_ + c)*HS_ + dd]);
        return;
    }
    rem -= C_*768;
    if (rem < C_*C_) {                        // projwT[d][c] = Wp[c][d]
        int d = rem / C_, c = rem % C_;
        projwT[(long)l*C_*C_ + rem] = f2bf(Wp[((long)l*C_ + c)*C_ + d]);
        return;
    }
    rem -= C_*C_;
    if (rem < FFN_*C_) {                      // w1T[f][c] = W1[c][f]
        int f = rem / C_, c = rem % C_;
        w1T[(long)l*FFN_*C_ + rem] = f2bf(W1[((long)l*C_ + c)*FFN_ + f]);
        return;
    }
    rem -= FFN_*C_;
    {                                         // w2T[co][f] = W2[f][co]
        int co = rem / FFN_, f = rem % FFN_;
        w2T[(long)l*C_*FFN_ + rem] = f2bf(W2[((long)l*FFN_ + f)*C_ + co]);
    }
}

// ---------------- embedding ----------------
__global__ __launch_bounds__(256) void embed_k(const int* __restrict__ idx,
                                               const float* __restrict__ tok,
                                               const float* __restrict__ pos,
                                               float* __restrict__ x)
{
    size_t i = (size_t)blockIdx.x * 256 + threadIdx.x;   // over B*T*C
    int c = (int)(i & (C_-1));
    size_t bt = i >> 8;
    size_t t  = bt & (T_-1);
    x[i] = tok[(size_t)idx[bt]*C_ + c] + pos[t*C_ + c];
}

// ---------------- layernorm (one block of 256 per row) ----------------
__global__ __launch_bounds__(256) void layernorm_k(const float* __restrict__ x,
                                                   const float* __restrict__ g,
                                                   const float* __restrict__ bta,
                                                   float* __restrict__ hf,
                                                   __bf16* __restrict__ hb)
{
    __shared__ float red[256];
    const int tid = threadIdx.x;
    const size_t row = blockIdx.x;
    float v = x[row*C_ + tid];

    red[tid] = v; __syncthreads();
    for (int s = 128; s > 0; s >>= 1) { if (tid < s) red[tid] += red[tid+s]; __syncthreads(); }
    float mean = red[0] * (1.0f/C_); __syncthreads();

    float d = v - mean;
    red[tid] = d*d; __syncthreads();
    for (int s = 128; s > 0; s >>= 1) { if (tid < s) red[tid] += red[tid+s]; __syncthreads(); }
    float rs = rsqrtf(red[0] * (1.0f/C_) + EPS_);

    float o = d * rs * g[tid] + bta[tid];
    if (hf) hf[row*C_ + tid] = o;
    if (hb) hb[row*C_ + tid] = f2bf(o);
}

// ---------------- generic bf16 WMMA GEMM, B given transposed ----------------
// C[M,N] = A[M,K](bf16) * BT[N,K](bf16)^T (+bias)(relu)(+resid) -> fp32 and/or bf16
// block 256 threads = 8 waves (2 M x 4 N), wave 32x32 via 2x2 16x16 frags,
// K-tile 64, double-buffered LDS with async global->LDS copies when available.
__global__ __launch_bounds__(256) void gemm_bf16_wmma(
    const __bf16* __restrict__ A, int lda,
    const __bf16* __restrict__ BT, int ldbt,
    const float* __restrict__ bias,
    const float* __restrict__ resid, int ldr,
    float* __restrict__ Cf, int ldcf,
    __bf16* __restrict__ Cb, int ldcb,
    int K, int relu)
{
    __shared__ __bf16 As [2][64][72];     // 64 rows x 64 k  (+8 pad, 144B pitch)
    __shared__ __bf16 BsT[2][128][72];    // 128 n   x 64 k  (+8 pad)

    const int tid  = threadIdx.x;
    const int lane = tid & 31;
    const int wid  = tid >> 5;
    const int wm   = wid >> 2;        // 0..1
    const int wn   = wid & 3;        // 0..3
    const int lr   = lane & 15;
    const int kbase = (lane >> 4) * 8;
    const int half = lane >> 4;
    const int blockM = blockIdx.y * 64;
    const int blockN = blockIdx.x * 128;

    v8f acc[2][2];
    #pragma unroll
    for (int a = 0; a < 2; a++)
      #pragma unroll
      for (int b = 0; b < 2; b++)
        #pragma unroll
        for (int i = 0; i < 8; i++) acc[a][b][i] = 0.0f;

    const int ar = tid >> 2, ac = (tid & 3) * 16;   // A staging coords
    const int bn = tid >> 1, bc = (tid & 1) * 32;   // B staging coords

    auto stage = [&](int buf, int k0) {
        const __bf16* ga = &A [(size_t)(blockM + ar) * lda  + k0 + ac];
        const __bf16* gb = &BT[(size_t)(blockN + bn) * ldbt + k0 + bc];
#if GEMM_ASYNC
        async_cp16(ga,      &As[buf][ar][ac]);
        async_cp16(ga + 8,  &As[buf][ar][ac + 8]);
        async_cp16(gb,      &BsT[buf][bn][bc]);
        async_cp16(gb + 8,  &BsT[buf][bn][bc + 8]);
        async_cp16(gb + 16, &BsT[buf][bn][bc + 16]);
        async_cp16(gb + 24, &BsT[buf][bn][bc + 24]);
#else
        uint4 a0 = ((const uint4*)ga)[0];
        uint4 a1 = ((const uint4*)ga)[1];
        uint4 b0 = ((const uint4*)gb)[0];
        uint4 b1 = ((const uint4*)gb)[1];
        uint4 b2 = ((const uint4*)gb)[2];
        uint4 b3 = ((const uint4*)gb)[3];
        uint4* da = (uint4*)&As[buf][ar][ac];
        da[0] = a0; da[1] = a1;
        uint4* db = (uint4*)&BsT[buf][bn][bc];
        db[0] = b0; db[1] = b1; db[2] = b2; db[3] = b3;
#endif
    };

    stage(0, 0);
    int buf = 0;
    for (int k0 = 0; k0 < K; k0 += 64) {
#if GEMM_ASYNC
        wait_async0();                       // current buf's copies complete
#endif
        __syncthreads();                     // whole tile visible to all waves
        if (k0 + 64 < K) stage(buf ^ 1, k0 + 64);   // overlap next tile copy

        #pragma unroll
        for (int ks = 0; ks < 2; ks++) {
            v16bf afrag[2], bfrag[2];
            #pragma unroll
            for (int fm = 0; fm < 2; fm++)
                afrag[fm] = load_frag(&As[buf][wm*32 + fm*16 + lr][ks*32 + kbase]);
            #pragma unroll
            for (int fn = 0; fn < 2; fn++)
                bfrag[fn] = load_frag(&BsT[buf][wn*32 + fn*16 + lr][ks*32 + kbase]);
            #pragma unroll
            for (int fm = 0; fm < 2; fm++)
              #pragma unroll
              for (int fn = 0; fn < 2; fn++)
                acc[fm][fn] = __builtin_amdgcn_wmma_f32_16x16x32_bf16(
                    false, afrag[fm], false, bfrag[fn], (short)0, acc[fm][fn], false, false);
        }
        buf ^= 1;
    }

    // fused epilogue
    #pragma unroll
    for (int fm = 0; fm < 2; fm++) {
      #pragma unroll
      for (int fn = 0; fn < 2; fn++) {
        int col = blockN + wn*32 + fn*16 + lr;
        float bv = bias ? bias[col] : 0.0f;
        #pragma unroll
        for (int r = 0; r < 8; r++) {
            int row = blockM + wm*32 + fm*16 + half*8 + r;
            float v = acc[fm][fn][r] + bv;
            if (relu) v = fmaxf(v, 0.0f);
            if (resid) v += resid[(size_t)row * ldr + col];
            if (Cf) Cf[(size_t)row * ldcf + col] = v;
            if (Cb) Cb[(size_t)row * ldcb + col] = f2bf(v);
        }
      }
    }
}

// ---------------- flash attention ----------------
// qkv [NTOK,768] bf16 (Q|K|V per head).  One wave = 16 query rows, block = 128 rows.
// grid: (T/128, B*H)
__global__ __launch_bounds__(256) void attn_flash_k(const __bf16* __restrict__ qkv,
                                                    __bf16* __restrict__ attn_out,
                                                    float scale)
{
    __shared__ __bf16 Kt [32][40];     // [key][c]   80B pitch
    __shared__ __bf16 VtT[32][40];     // [d][key]   transposed on store
    __shared__ __bf16 Plds[8][16][32]; // per-wave P tile, 64B pitch

    const int tid  = threadIdx.x;
    const int lane = tid & 31;
    const int w    = tid >> 5;
    const int lr   = lane & 15;
    const int half = lane >> 4;
    const int kbase = half * 8;
    const int bh = blockIdx.y;
    const int b  = bh / H_, h = bh % H_;
    const int q0 = blockIdx.x * 128 + w * 16;

    // Q fragment (A-matrix 16x32) straight from global: two b128 loads
    v16bf qf = load_frag(&qkv[(size_t)(b*T_ + q0 + lr) * 768 + h*HS_ + kbase]);

    v8f o0, o1;
    float mst[8], lst[8];
    #pragma unroll
    for (int r = 0; r < 8; r++) { o0[r] = 0.f; o1[r] = 0.f; mst[r] = -3.0e38f; lst[r] = 0.f; }

    const int sr = tid >> 3, sc = (tid & 7) * 4;     // staging coords

    for (int kb = 0; kb < T_; kb += 32) {
        __syncthreads();
        {   // stage K (async if available) and V (transposed through regs)
            size_t base = (size_t)(b*T_ + kb + sr) * 768 + h*HS_ + sc;
#if ATTN_ASYNC
            async_cp8(&qkv[base + 256], &Kt[sr][sc]);
#else
            *(uint2*)&Kt[sr][sc] = *(const uint2*)&qkv[base + 256];
#endif
            union { uint2 u; __bf16 e[4]; } vv;
            vv.u = *(const uint2*)&qkv[base + 512];
            #pragma unroll
            for (int j = 0; j < 4; j++) VtT[sc + j][sr] = vv.e[j];
            if (kb + 32 < T_)
                __builtin_prefetch(&qkv[base + 256 + (size_t)32*768], 0, 1);
        }
#if ATTN_ASYNC
        wait_async0();
#endif
        __syncthreads();

        // S = Q * K^T : B[k=c][n=key] = Kt[key][c]
        v16bf bk0 = load_frag(&Kt[lr][kbase]);
        v16bf bk1 = load_frag(&Kt[lr + 16][kbase]);
        v8f s0, s1;
        #pragma unroll
        for (int r = 0; r < 8; r++) { s0[r] = 0.f; s1[r] = 0.f; }
        s0 = __builtin_amdgcn_wmma_f32_16x16x32_bf16(false, qf, false, bk0, (short)0, s0, false, false);
        s1 = __builtin_amdgcn_wmma_f32_16x16x32_bf16(false, qf, false, bk1, (short)0, s1, false, false);

        // online softmax over the 32 new keys
        float mnew[8], al[8];
        #pragma unroll
        for (int r = 0; r < 8; r++) {
            float a = s0[r] * scale, c = s1[r] * scale;
            s0[r] = a; s1[r] = c;
            float mx = fmaxf(a, c);
            #pragma unroll
            for (int off = 1; off < 16; off <<= 1) mx = fmaxf(mx, __shfl_xor(mx, off, 32));
            mnew[r] = fmaxf(mst[r], mx);
            al[r] = __expf(mst[r] - mnew[r]);
            mst[r] = mnew[r];
        }
        #pragma unroll
        for (int r = 0; r < 8; r++) {
            float p0 = __expf(s0[r] - mnew[r]);
            float p1 = __expf(s1[r] - mnew[r]);
            float rs = p0 + p1;
            #pragma unroll
            for (int off = 1; off < 16; off <<= 1) rs += __shfl_xor(rs, off, 32);
            lst[r] = lst[r] * al[r] + rs;
            o0[r] *= al[r]; o1[r] *= al[r];
            Plds[w][half*8 + r][lr]      = f2bf(p0);
            Plds[w][half*8 + r][lr + 16] = f2bf(p1);
        }

        // O += P * V : P as A-fragment via per-wave LDS; V^T gives contiguous B-frags
        v16bf pf  = load_frag(&Plds[w][lr][kbase]);
        v16bf bv0 = load_frag(&VtT[lr][kbase]);
        v16bf bv1 = load_frag(&VtT[lr + 16][kbase]);
        o0 = __builtin_amdgcn_wmma_f32_16x16x32_bf16(false, pf, false, bv0, (short)0, o0, false, false);
        o1 = __builtin_amdgcn_wmma_f32_16x16x32_bf16(false, pf, false, bv1, (short)0, o1, false, false);
    }

    #pragma unroll
    for (int r = 0; r < 8; r++) {
        float inv = 1.0f / lst[r];
        size_t row = (size_t)(b*T_ + q0 + half*8 + r) * C_ + h*HS_;
        attn_out[row + lr]      = f2bf(o0[r] * inv);
        attn_out[row + lr + 16] = f2bf(o1[r] * inv);
    }
}

// ---------------- mean pool over T ----------------
__global__ __launch_bounds__(256) void meanpool_k(const float* __restrict__ hf,
                                                  float* __restrict__ emb)
{
    int id = blockIdx.x * 256 + threadIdx.x;   // 0..B*C-1
    int b = id >> 8, c = id & 255;
    float s = 0.f;
    for (int t = 0; t < T_; t++) s += hf[((size_t)(b*T_ + t))*C_ + c];
    emb[id] = s * (1.0f / T_);
}

// ---------------- classifier + softmax (tiny, one block) ----------------
__global__ __launch_bounds__(512) void classifier_k(const float* __restrict__ emb,
                                                    const float* __restrict__ Wc1,
                                                    const float* __restrict__ bc1,
                                                    const float* __restrict__ Wc2,
                                                    const float* __restrict__ bc2,
                                                    float* __restrict__ out)
{
    __shared__ float se[B_*C_];
    __shared__ float sh[B_*CLSH_];
    __shared__ float lg[B_*NOUT_];
    int tid = threadIdx.x;
    for (int i = tid; i < B_*C_; i += 512) se[i] = emb[i];
    __syncthreads();
    for (int b = 0; b < B_; b++) {
        float a = bc1[tid];
        for (int c = 0; c < C_; c++) a += se[b*C_ + c] * Wc1[(size_t)c*CLSH_ + tid];
        sh[b*CLSH_ + tid] = fmaxf(a, 0.f);
    }
    __syncthreads();
    if (tid < B_*NOUT_) {
        int b = tid / NOUT_, o = tid % NOUT_;
        float a = bc2[o];
        for (int j = 0; j < CLSH_; j++) a += sh[b*CLSH_ + j] * Wc2[(size_t)j*NOUT_ + o];
        lg[tid] = a;
    }
    __syncthreads();
    if (tid < B_) {
        float mx = -1e30f;
        for (int o = 0; o < NOUT_; o++) mx = fmaxf(mx, lg[tid*NOUT_ + o]);
        float e[NOUT_], s = 0.f;
        for (int o = 0; o < NOUT_; o++) { e[o] = __expf(lg[tid*NOUT_ + o] - mx); s += e[o]; }
        float inv = 1.0f / s;
        for (int o = 0; o < NOUT_; o++) out[tid*NOUT_ + o] = e[o] * inv;
    }
}

// ---------------- host orchestration ----------------
extern "C" void kernel_launch(void* const* d_in, const int* in_sizes, int n_in,
                              void* d_out, int out_size, void* d_ws, size_t ws_size,
                              hipStream_t stream)
{
    const int*   idx    = (const int*)  d_in[0];
    const float* tok    = (const float*)d_in[1];
    const float* pos    = (const float*)d_in[2];
    const float* Wq     = (const float*)d_in[3];
    const float* Wk     = (const float*)d_in[4];
    const float* Wv     = (const float*)d_in[5];
    const float* Wproj  = (const float*)d_in[6];
    const float* bproj  = (const float*)d_in[7];
    const float* ln1_g  = (const float*)d_in[8];
    const float* ln1_b  = (const float*)d_in[9];
    const float* ln2_g  = (const float*)d_in[10];
    const float* ln2_b  = (const float*)d_in[11];
    const float* W1     = (const float*)d_in[12];
    const float* b1     = (const float*)d_in[13];
    const float* W2     = (const float*)d_in[14];
    const float* b2     = (const float*)d_in[15];
    const float* lnf_g  = (const float*)d_in[16];
    const float* lnf_b  = (const float*)d_in[17];
    const float* Wc1    = (const float*)d_in[18];
    const float* bc1    = (const float*)d_in[19];
    const float* Wc2    = (const float*)d_in[20];
    const float* bc2    = (const float*)d_in[21];
    float* out = (float*)d_out;

    // ---- workspace carve-up (256B aligned) ----
    char* p = (char*)d_ws;
    auto alloc = [&](size_t bytes) -> void* {
        void* q = (void*)p;
        p += (bytes + 255) & ~(size_t)255;
        return q;
    };
    float*  x      = (float*) alloc((size_t)NTOK_*C_*4);     // residual stream fp32
    float*  hf     = (float*) alloc((size_t)NTOK_*C_*4);     // LN fp32 out (final LN / pool)
    __bf16* hb     = (__bf16*)alloc((size_t)NTOK_*C_*2);     // LN bf16 out
    __bf16* qkv    = (__bf16*)alloc((size_t)NTOK_*768*2);    // fused QKV
    __bf16* attn   = (__bf16*)alloc((size_t)NTOK_*C_*2);     // attention output (heads concat)
    __bf16* fbuf   = (__bf16*)alloc((size_t)NTOK_*FFN_*2);   // FFN hidden
    __bf16* qkvwT  = (__bf16*)alloc((size_t)L_*768*C_*2);
    __bf16* projwT = (__bf16*)alloc((size_t)L_*C_*C_*2);
    __bf16* w1T    = (__bf16*)alloc((size_t)L_*FFN_*C_*2);
    __bf16* w2T    = (__bf16*)alloc((size_t)L_*C_*FFN_*2);
    float*  emb    = (float*) alloc((size_t)B_*C_*4);

    // 1. pack (and transpose) weights to bf16
    {
        long total = (long)L_*(C_*768 + C_*C_ + C_*FFN_ + FFN_*C_);
        int blocks = (int)((total + 255) / 256);
        pack_weights_k<<<blocks, 256, 0, stream>>>(Wq, Wk, Wv, Wproj, W1, W2,
                                                   qkvwT, projwT, w1T, w2T);
    }
    // 2. embedding
    embed_k<<<(NTOK_*C_)/256, 256, 0, stream>>>(idx, tok, pos, x);

    const float scale = 1.0f / 16.0f;    // C^-0.5 (faithful to reference)
    for (int l = 0; l < L_; l++) {
        // LN1
        layernorm_k<<<NTOK_, 256, 0, stream>>>(x, ln1_g + l*C_, ln1_b + l*C_, nullptr, hb);
        // fused QKV GEMM: [8192,256] x [256,768] -> bf16 qkv
        gemm_bf16_wmma<<<dim3(768/128, NTOK_/64), 256, 0, stream>>>(
            hb, C_, qkvwT + (size_t)l*768*C_, C_,
            nullptr, nullptr, 0, nullptr, 0, qkv, 768, C_, 0);
        // flash attention
        attn_flash_k<<<dim3(T_/128, B_*H_), 256, 0, stream>>>(qkv, attn, scale);
        // proj GEMM + bias + residual into x
        gemm_bf16_wmma<<<dim3(C_/128, NTOK_/64), 256, 0, stream>>>(
            attn, C_, projwT + (size_t)l*C_*C_, C_,
            bproj + l*C_, x, C_, x, C_, nullptr, 0, C_, 0);
        // LN2
        layernorm_k<<<NTOK_, 256, 0, stream>>>(x, ln2_g + l*C_, ln2_b + l*C_, nullptr, hb);
        // FFN1 + bias + relu -> bf16
        gemm_bf16_wmma<<<dim3(FFN_/128, NTOK_/64), 256, 0, stream>>>(
            hb, C_, w1T + (size_t)l*FFN_*C_, C_,
            b1 + l*FFN_, nullptr, 0, nullptr, 0, fbuf, FFN_, C_, 1);
        // FFN2 + bias + residual into x
        gemm_bf16_wmma<<<dim3(C_/128, NTOK_/64), 256, 0, stream>>>(
            fbuf, FFN_, w2T + (size_t)l*C_*FFN_, FFN_,
            b2 + l*C_, x, C_, x, C_, nullptr, 0, FFN_, 0);
    }
    // final LN (fp32 out), mean pool, classifier+softmax
    layernorm_k<<<NTOK_, 256, 0, stream>>>(x, lnf_g, lnf_b, hf, nullptr);
    meanpool_k<<<(B_*C_)/256, 256, 0, stream>>>(hf, emb);
    classifier_k<<<1, 512, 0, stream>>>(emb, Wc1, bc1, Wc2, bc2, out);
}
